// HeteroGATConv_9869834846328
// MI455X (gfx1250) — compile-verified, matrix operand order checked by reading, not run on hardware
//
#include <hip/hip_runtime.h>
#include <hip/hip_bf16.h>

#define NN   100000
#define EE   1600000
#define FDIM 128
#define HH   4
#define DD   32
#define LL   3
#define SLOPE 0.2f

typedef __attribute__((ext_vector_type(16))) __bf16          v16bf;
typedef __attribute__((ext_vector_type(16))) unsigned short  v16us;
typedef __attribute__((ext_vector_type(8)))  float           v8f;

__device__ __forceinline__ unsigned short f2bf(float f) {
    union { float f; unsigned int u; } c; c.f = f;
    unsigned int u = c.u;
    u += 0x7FFFu + ((u >> 16) & 1u);      // round-to-nearest-even
    return (unsigned short)(u >> 16);
}

// ---- operand pre-pack ------------------------------------------------------
// Within each 32-column group, permute columns [0..7,16..23,8..15,24..31] so a
// lane's 16 A-values (K = khalf+[0,8) and 16+khalf+[0,8)) are 32 contiguous
// bytes. The permutation is an involution: read permuted, write coalesced.
__global__ void gat_pack_x(const float* __restrict__ X, unsigned short* __restrict__ XBF) {
    int i = blockIdx.x * blockDim.x + threadIdx.x;       // over N*128
    if (i >= NN * FDIM) return;
    int col = i & (FDIM - 1);
    int kk  = col & ~31;
    int j   = col & 31;
    int g   = j >> 3;
    int gp  = (g == 1) ? 2 : (g == 2) ? 1 : g;           // swap middle groups
    int p   = (j & 7) | (gp << 3);
    XBF[i] = f2bf(X[(i & ~(FDIM - 1)) | (kk + p)]);
}

// WBF[nt][ks][lane][v] : lane's 16 B-values contiguous. W row-major [K,Nout].
__global__ void gat_pack_w(const float* __restrict__ W, unsigned short* __restrict__ WBF) {
    int i = blockIdx.x * blockDim.x + threadIdx.x;       // 8*4*32*16 = 16384
    if (i >= 16384) return;
    int v    = i & 15;
    int lane = (i >> 4) & 31;
    int ks   = (i >> 9) & 3;
    int nt   = i >> 11;
    int khalf = (lane >> 4) * 8;
    int k = ks * 32 + khalf + ((v < 8) ? v : 8 + v);     // v>=8 -> 16+khalf+(v-8)
    int n = nt * 16 + (lane & 15);
    WBF[i] = f2bf(W[k * FDIM + n]);
}

// ---- ft = X @ W via bf16 WMMA, f32 accumulate ------------------------------
// Block = 256 threads = 8 waves; block mt owns rows [mt*16,mt*16+16);
// wave wv owns cols [wv*16,wv*16+16). Hot loop: 2x b128-pair loads + 1 wmma.
__global__ void __launch_bounds__(256) gat_gemm_wmma(
    const unsigned short* __restrict__ XBF, const unsigned short* __restrict__ WBF,
    float* __restrict__ FT) {
    const int lane = threadIdx.x & 31;
    const int wv   = threadIdx.x >> 5;
    const int mt   = blockIdx.x;
    const int l15  = lane & 15;
    const int hi   = lane >> 4;
    v8f acc = {};
    const unsigned short* ap = XBF + (long)(mt * 16 + l15) * FDIM + hi * 16;
    const unsigned short* bp = WBF + ((wv * 4) * 32 + lane) * 16;
    #pragma unroll
    for (int ks = 0; ks < 4; ++ks) {
        union { v16bf v; v16us u; } A, B;
        A.u = *(const v16us*)(ap + ks * 32);
        B.u = *(const v16us*)(bp + ks * 32 * 16);
        acc = __builtin_amdgcn_wmma_f32_16x16x32_bf16(
                  false, A.v, false, B.v, (short)0, acc, false, false);
    }
    // D layout: VGPR i -> M = i + hi*8, N = lane%16
    float* o = FT + (long)(mt * 16 + hi * 8) * FDIM + wv * 16 + l15;
    #pragma unroll
    for (int i = 0; i < 8; ++i) o[(long)i * FDIM] = acc[i];
}

// ---- attention projections + init ------------------------------------------
__global__ void gat_proj_init(const float* __restrict__ FT,
                              const float* __restrict__ al, const float* __restrict__ ar,
                              float* __restrict__ el, float* __restrict__ er,
                              unsigned int* __restrict__ menc, float* __restrict__ den) {
    int idx = blockIdx.x * blockDim.x + threadIdx.x;
    if (idx >= NN * HH) return;
    int n = idx >> 2, h = idx & 3;
    const float* f = FT + (long)n * FDIM + h * DD;
    const float* L = al + h * DD;
    const float* R = ar + h * DD;
    float sl = 0.f, sr = 0.f;
    #pragma unroll
    for (int d = 0; d < DD; ++d) { float v = f[d]; sl += v * L[d]; sr += v * R[d]; }
    el[idx] = sl; er[idx] = sr;
    menc[idx] = 0u;       // below enc(-inf) -> decodes to NaN -> m=0 guard
    den[idx]  = 0.f;
}

__device__ __forceinline__ float leaky(float x) { return x > 0.f ? x : SLOPE * x; }

__device__ __forceinline__ unsigned int fenc(float x) {
    unsigned int u = __float_as_uint(x);
    return (u & 0x80000000u) ? ~u : (u | 0x80000000u);
}

// segment max of edge logits into dst nodes (order-preserving uint encoding)
__global__ void gat_edge_max(const int* __restrict__ src, const int* __restrict__ dst,
                             const float* __restrict__ el, const float* __restrict__ er,
                             unsigned int* __restrict__ menc) {
    int idx = blockIdx.x * blockDim.x + threadIdx.x;
    if (idx >= EE * HH) return;
    int e = idx >> 2, h = idx & 3;
    int s = src[e], d = dst[e];
    float x = leaky(el[s * HH + h] + er[d * HH + h]);
    atomicMax(&menc[d * HH + h], fenc(x));
}

// decode encoded max in place; non-finite (incl. untouched) -> 0
__global__ void gat_m_decode(unsigned int* __restrict__ menc, float* __restrict__ m) {
    int idx = blockIdx.x * blockDim.x + threadIdx.x;
    if (idx >= NN * HH) return;
    unsigned int u = menc[idx];
    unsigned int bits = (u & 0x80000000u) ? (u & 0x7FFFFFFFu) : ~u;
    float v = __uint_as_float(bits);
    if (((bits >> 23) & 0xFFu) == 0xFFu) v = 0.f;   // inf/NaN guard
    m[idx] = v;
}

// rst = residual + bias (scatter accumulates on top)
__global__ void gat_rst_init(const float* __restrict__ xc, const float* __restrict__ bias,
                             float* __restrict__ rst) {
    int i = blockIdx.x * blockDim.x + threadIdx.x;
    if (i >= NN * FDIM) return;
    rst[i] = xc[i] + bias[i & (FDIM - 1)];
}

// ee = exp(e - m[dst]); store into a-buffer; accumulate denominator
__global__ void gat_edge_exp(const int* __restrict__ src, const int* __restrict__ dst,
                             const float* __restrict__ el, const float* __restrict__ er,
                             const float* __restrict__ m, float* __restrict__ den,
                             float* __restrict__ aout) {
    int idx = blockIdx.x * blockDim.x + threadIdx.x;
    if (idx >= EE * HH) return;
    int e = idx >> 2, h = idx & 3;
    int s = src[e], d = dst[e];
    float x  = leaky(el[s * HH + h] + er[d * HH + h]);
    float ee = __expf(x - m[d * HH + h]);
    aout[idx] = ee;
    atomicAdd(&den[d * HH + h], ee);
}

// a = ee/den; rst[dst] += ft[src]*a. One wave per (edge, head); lane = feature.
__global__ void __launch_bounds__(256) gat_scatter(
    const int* __restrict__ src, const int* __restrict__ dst,
    const float* __restrict__ FT, const float* __restrict__ den,
    float* __restrict__ aout, float* __restrict__ rst) {
    int t = blockIdx.x * 256 + threadIdx.x;
    int e = t >> 7;          // 128 threads per edge
    int i = t & (FDIM - 1);  // feature index, head = i/32
    if (e >= EE) return;
    int h = i >> 5;
    int s = src[e], d = dst[e];
    float ee = aout[e * HH + h];
    float a  = ee / fmaxf(den[d * HH + h], 1e-9f);
    if ((i & 31) == 0) aout[e * HH + h] = a;   // same wave read-then-write: safe
    atomicAdd(&rst[(long)d * FDIM + i], FT[(long)s * FDIM + i] * a);
}

__global__ void gat_elu(const float* __restrict__ rst, float* __restrict__ xn) {
    int i = blockIdx.x * blockDim.x + threadIdx.x;
    if (i >= NN * FDIM) return;
    float v = rst[i];
    xn[i] = v > 0.f ? v : (__expf(v) - 1.0f);
}

extern "C" void kernel_launch(void* const* d_in, const int* in_sizes, int n_in,
                              void* d_out, int out_size, void* d_ws, size_t ws_size,
                              hipStream_t stream) {
    const float* h    = (const float*)d_in[0];
    const int*   src  = (const int*)d_in[1];
    const int*   dst  = (const int*)d_in[2];
    const float* Ws   = (const float*)d_in[3];
    const float* al   = (const float*)d_in[4];
    const float* ar   = (const float*)d_in[5];
    const float* bias = (const float*)d_in[6];

    float* out   = (float*)d_out;                 // [N,128] final features
    float* a_out = out + (size_t)NN * FDIM;       // [E,H] attention output/scratch

    float* ft  = (float*)d_ws;                    // [N,128]
    float* rst = ft  + (size_t)NN * FDIM;         // [N,128]
    float* el  = rst + (size_t)NN * FDIM;         // [N,H]
    float* er  = el  + (size_t)NN * HH;           // [N,H]
    float* m   = er  + (size_t)NN * HH;           // [N,H] uint enc -> float
    float* den = m   + (size_t)NN * HH;           // [N,H]
    unsigned short* xbf = (unsigned short*)(den + (size_t)NN * HH);  // [N,128] bf16 packed
    unsigned short* wbf = xbf + (size_t)NN * FDIM;                   // [16384] bf16 packed

    const int nh_blocks = (NN * HH + 255) / 256;
    const int eh_blocks = (EE * HH + 255) / 256;
    const int nf_blocks = (NN * FDIM + 255) / 256;

    for (int l = 0; l < LL; ++l) {
        const float* xc = (l == 0) ? h : out;     // d_out head doubles as x buffer
        gat_pack_w<<<16384 / 256, 256, 0, stream>>>(Ws + (size_t)l * FDIM * FDIM, wbf);
        gat_pack_x<<<nf_blocks, 256, 0, stream>>>(xc, xbf);
        gat_gemm_wmma<<<NN / 16, 256, 0, stream>>>(xbf, wbf, ft);
        gat_proj_init<<<nh_blocks, 256, 0, stream>>>(ft, al + l * HH * DD, ar + l * HH * DD,
                                                     el, er, (unsigned int*)m, den);
        gat_edge_max<<<eh_blocks, 256, 0, stream>>>(src, dst, el, er, (unsigned int*)m);
        gat_m_decode<<<nh_blocks, 256, 0, stream>>>((unsigned int*)m, m);
        gat_rst_init<<<nf_blocks, 256, 0, stream>>>(xc, bias + l * FDIM, rst);
        gat_edge_exp<<<eh_blocks, 256, 0, stream>>>(src, dst, el, er, m, den, a_out);
        gat_scatter<<<(EE * FDIM) / 256, 256, 0, stream>>>(src, dst, ft, den, a_out, rst);
        gat_elu<<<nf_blocks, 256, 0, stream>>>(rst, out);
    }
}
